// GCNIIBackbone_12695923327658
// MI455X (gfx1250) — compile-verified, matrix operand order checked by reading, not run on hardware
//
#include <hip/hip_runtime.h>
#include <hip/hip_bf16.h>
#include <math.h>

// ---------------------------------------------------------------------------
// Types for CDNA5 WMMA (wave32): v_wmma_f32_16x16x32_bf16
// ---------------------------------------------------------------------------
typedef __attribute__((ext_vector_type(16))) __bf16 v16bf;
typedef __attribute__((ext_vector_type(8)))  __bf16 v8bf;
typedef __attribute__((ext_vector_type(8)))  float  v8f;

#define DHID 256
#define LDA  (DHID + 8)   // padded LDS row stride (bf16): 528B -> 4-bank shift/row

// ---------------------------------------------------------------------------
// Fragment loader per ISA 7.12.2 (16-bit A 16x32 layout):
//   lanes 0-15 : row M = lane,    K = kbase+[0..7]  and kbase+[16..23]
//   lanes 16-31: row M = lane-16, K = kbase+[8..15] and kbase+[24..31]
// Works for global (W^T row-major [N][K]) and LDS-staged tiles alike.
// ---------------------------------------------------------------------------
__device__ __forceinline__ v16bf frag_bf16(const __bf16* base, int ld, int kbase) {
    int lane = threadIdx.x & 31;
    int r    = lane & 15;
    int koff = (lane < 16) ? 0 : 8;
    const __bf16* p = base + (size_t)r * ld + kbase + koff;
    v8bf lo = *reinterpret_cast<const v8bf*>(p);
    v8bf hi = *reinterpret_cast<const v8bf*>(p + 16);
    v16bf f;
#pragma unroll
    for (int i = 0; i < 8; ++i) { f[i] = lo[i]; f[i + 8] = hi[i]; }
    return f;
}

__device__ __forceinline__ v8f wmma_bf16(v16bf a, v16bf b, v8f c) {
    return __builtin_amdgcn_wmma_f32_16x16x32_bf16(false, a, false, b, (short)0, c, false, false);
}

__device__ __forceinline__ v8bf cvt8(float4 a, float4 b) {
    v8bf o;
    o[0] = (__bf16)a.x; o[1] = (__bf16)a.y; o[2] = (__bf16)a.z; o[3] = (__bf16)a.w;
    o[4] = (__bf16)b.x; o[5] = (__bf16)b.y; o[6] = (__bf16)b.z; o[7] = (__bf16)b.w;
    return o;
}

// ---------------------------------------------------------------------------
// Degree / normalization kernels
// ---------------------------------------------------------------------------
__global__ void k_deg(const int* __restrict__ col, float* __restrict__ deg, int E) {
    int e = blockIdx.x * blockDim.x + threadIdx.x;
    if (e < E) atomicAdd(deg + col[e], 1.0f);
}

__global__ void k_dis(float* __restrict__ deg, int N) {
    int i = blockIdx.x * blockDim.x + threadIdx.x;
    if (i < N) deg[i] = rsqrtf(deg[i] + 1.0f);   // +1 accounts for the self loop
}

__global__ void k_enorm(const int* __restrict__ row, const int* __restrict__ col,
                        const float* __restrict__ dis, float* __restrict__ enorm, int E) {
    int e = blockIdx.x * blockDim.x + threadIdx.x;
    if (e < E) enorm[e] = dis[row[e]] * dis[col[e]];
}

// ---------------------------------------------------------------------------
// Weight conversion: fp32 [K][N] -> bf16 W^T [N][K]
// ---------------------------------------------------------------------------
__global__ void k_wt_bf16(const float* __restrict__ src, __bf16* __restrict__ dst,
                          int K, int Nn, int total) {
    int t = blockIdx.x * blockDim.x + threadIdx.x;
    if (t >= total) return;
    int mat = t / (K * Nn);
    int r   = t - mat * K * Nn;
    int n   = r / K;
    int k   = r - n * K;
    dst[t]  = (__bf16)src[(size_t)mat * K * Nn + (size_t)k * Nn + n];
}

// ---------------------------------------------------------------------------
// GEMM: C[M,Nout] = relu(A[M,K] @ W + bias), W given as bf16 W^T [Nout][K].
// A-tile (16 x K fp32) staged once per block into LDS as bf16.
// One wave -> 16x64 tile. blockDim.x = 32*(Nout/64), gridDim.x = M/16.
// ---------------------------------------------------------------------------
__global__ void k_gemm_bias_relu(const float* __restrict__ A, const __bf16* __restrict__ Wt,
                                 const float* __restrict__ bias, float* __restrict__ C,
                                 int K, int Nout) {
    __shared__ __attribute__((aligned(16))) __bf16 sA[16 * LDA];

    int wave = threadIdx.x >> 5;
    int lane = threadIdx.x & 31;
    int m0   = blockIdx.x * 16;
    int n0   = wave * 64;
    int lda  = K + 8;

    // cooperative stage: 16 rows x K cols, chunks of 8 floats
    int cpr = K >> 3;                       // chunks per row
    for (int t = threadIdx.x; t < 16 * cpr; t += blockDim.x) {
        int row = t / cpr;
        int cc  = (t - row * cpr) * 8;
        const float4* p = reinterpret_cast<const float4*>(A + (size_t)(m0 + row) * K + cc);
        *reinterpret_cast<v8bf*>(sA + row * lda + cc) = cvt8(p[0], p[1]);
    }
    __syncthreads();

    v8f acc[4];
#pragma unroll
    for (int t = 0; t < 4; ++t)
#pragma unroll
        for (int j = 0; j < 8; ++j) acc[t][j] = 0.0f;

    for (int k = 0; k < K; k += 32) {
        v16bf a = frag_bf16(sA, lda, k);
#pragma unroll
        for (int t = 0; t < 4; ++t) {
            v16bf b = frag_bf16(Wt + (size_t)(n0 + t * 16) * K, K, k);
            acc[t] = wmma_bf16(a, b, acc[t]);
        }
    }

    int rbase = (lane < 16) ? 0 : 8;
    int cl    = lane & 15;
#pragma unroll
    for (int t = 0; t < 4; ++t) {
        int c = n0 + t * 16 + cl;
        float bb = bias[c];
#pragma unroll
        for (int v = 0; v < 8; ++v) {
            int r = m0 + rbase + v;
            C[(size_t)r * Nout + c] = fmaxf(acc[t][v] + bb, 0.0f);
        }
    }
}

// ---------------------------------------------------------------------------
// Sparse aggregation: agg[c] += w * h[r], one wave per (edge | self-loop).
// blockDim = 256 (8 edges/block); lane handles 8 consecutive features.
// ---------------------------------------------------------------------------
__global__ void k_agg(const int* __restrict__ row, const int* __restrict__ col,
                      const float* __restrict__ enorm, const float* __restrict__ dis,
                      const float* __restrict__ h, float* __restrict__ agg,
                      int E, int Nn) {
    int e = blockIdx.x * 8 + (threadIdx.x >> 5);
    if (e >= E + Nn) return;
    int lane = threadIdx.x & 31;
    int r, c; float w;
    if (e < E) { r = row[e]; c = col[e]; w = enorm[e]; }
    else       { r = c = e - E; float d = dis[r]; w = d * d; }

    const float4* src = reinterpret_cast<const float4*>(h + (size_t)r * DHID);
    float*        dst = agg + (size_t)c * DHID;
    float4 v0 = src[lane * 2 + 0];
    float4 v1 = src[lane * 2 + 1];
    int f = lane * 8;
    atomicAdd(dst + f + 0, w * v0.x); atomicAdd(dst + f + 1, w * v0.y);
    atomicAdd(dst + f + 2, w * v0.z); atomicAdd(dst + f + 3, w * v0.w);
    atomicAdd(dst + f + 4, w * v1.x); atomicAdd(dst + f + 5, w * v1.y);
    atomicAdd(dst + f + 6, w * v1.z); atomicAdd(dst + f + 7, w * v1.w);
}

// ---------------------------------------------------------------------------
// Fused dual GEMM + GCNII combine + LayerNorm statistics.
//   acc = agg @ W1 + x0 @ W2          (K = 256 each, bf16 WMMA)
//   h   = c0*(agg + x0) + c1*acc      (c0 = (1-beta)/2, c1 = beta/2)
//   stats[0] += sum(h); stats[1] += sum(h^2)
// agg/x0 tiles staged to LDS (bf16 fragments + exact fp32 skip-path sum).
// blockDim = 128 (4 waves cover N=256), gridDim.x = M/16.
// ---------------------------------------------------------------------------
__global__ void k_dualgemm_combine(const float* __restrict__ agg, const float* __restrict__ x0,
                                   const __bf16* __restrict__ w1t, const __bf16* __restrict__ w2t,
                                   float* __restrict__ h, float* __restrict__ stats,
                                   float c0, float c1) {
    __shared__ __attribute__((aligned(16))) __bf16 sAgg[16 * LDA];
    __shared__ __attribute__((aligned(16))) __bf16 sX0 [16 * LDA];
    __shared__ __attribute__((aligned(16))) float  sSum[16 * DHID];

    int wave = threadIdx.x >> 5;
    int lane = threadIdx.x & 31;
    int m0   = blockIdx.x * 16;
    int n0   = wave * 64;

    // cooperative stage: 16 rows x 256 cols; bf16 copies + fp32 sum
    for (int t = threadIdx.x; t < 16 * (DHID / 8); t += blockDim.x) {
        int row = t >> 5;
        int cc  = (t & 31) * 8;
        size_t goff = (size_t)(m0 + row) * DHID + cc;
        const float4* pa = reinterpret_cast<const float4*>(agg + goff);
        const float4* px = reinterpret_cast<const float4*>(x0 + goff);
        float4 a0 = pa[0], a1 = pa[1];
        float4 b0 = px[0], b1 = px[1];
        *reinterpret_cast<v8bf*>(sAgg + row * LDA + cc) = cvt8(a0, a1);
        *reinterpret_cast<v8bf*>(sX0  + row * LDA + cc) = cvt8(b0, b1);
        float4 s0 = make_float4(a0.x + b0.x, a0.y + b0.y, a0.z + b0.z, a0.w + b0.w);
        float4 s1 = make_float4(a1.x + b1.x, a1.y + b1.y, a1.z + b1.z, a1.w + b1.w);
        float4* ps = reinterpret_cast<float4*>(sSum + row * DHID + cc);
        ps[0] = s0; ps[1] = s1;
    }
    __syncthreads();

    v8f acc[4];
#pragma unroll
    for (int t = 0; t < 4; ++t)
#pragma unroll
        for (int j = 0; j < 8; ++j) acc[t][j] = 0.0f;

    for (int k = 0; k < DHID; k += 32) {
        v16bf a = frag_bf16(sAgg, LDA, k);
#pragma unroll
        for (int t = 0; t < 4; ++t) {
            v16bf b = frag_bf16(w1t + (size_t)(n0 + t * 16) * DHID, DHID, k);
            acc[t] = wmma_bf16(a, b, acc[t]);
        }
    }
    for (int k = 0; k < DHID; k += 32) {
        v16bf a = frag_bf16(sX0, LDA, k);
#pragma unroll
        for (int t = 0; t < 4; ++t) {
            v16bf b = frag_bf16(w2t + (size_t)(n0 + t * 16) * DHID, DHID, k);
            acc[t] = wmma_bf16(a, b, acc[t]);
        }
    }

    int rbase = (lane < 16) ? 0 : 8;
    int cl    = lane & 15;
    float ls = 0.0f, lss = 0.0f;
#pragma unroll
    for (int t = 0; t < 4; ++t) {
        int c = n0 + t * 16 + cl;
#pragma unroll
        for (int v = 0; v < 8; ++v) {
            int rl = rbase + v;
            float val = c0 * sSum[rl * DHID + c] + c1 * acc[t][v];
            h[(size_t)(m0 + rl) * DHID + c] = val;
            ls += val; lss += val * val;
        }
    }
#pragma unroll
    for (int o = 16; o > 0; o >>= 1) {
        ls  += __shfl_xor(ls,  o, 32);
        lss += __shfl_xor(lss, o, 32);
    }
    if (lane == 0) { atomicAdd(stats, ls); atomicAdd(stats + 1, lss); }
}

__global__ void k_ln_finalize(float* __restrict__ stats, float cnt) {
    float mean = stats[0] / cnt;
    float ms   = stats[1] / cnt - mean * mean;
    stats[2] = mean;
    stats[3] = 1.0f / (sqrtf(fmaxf(ms, 0.0f)) + 1e-5f);
}

__global__ void k_ln_apply(float* __restrict__ h, const float* __restrict__ stats,
                           const float* __restrict__ gamma, const float* __restrict__ beta,
                           int total4) {
    int i = blockIdx.x * blockDim.x + threadIdx.x;
    if (i >= total4) return;
    float mean = stats[2], inv = stats[3];
    float4 v = reinterpret_cast<float4*>(h)[i];
    int c = (i * 4) & (DHID - 1);
    v.x = fmaxf(gamma[c + 0] * ((v.x - mean) * inv) + beta[c + 0], 0.0f);
    v.y = fmaxf(gamma[c + 1] * ((v.y - mean) * inv) + beta[c + 1], 0.0f);
    v.z = fmaxf(gamma[c + 2] * ((v.z - mean) * inv) + beta[c + 2], 0.0f);
    v.w = fmaxf(gamma[c + 3] * ((v.w - mean) * inv) + beta[c + 3], 0.0f);
    reinterpret_cast<float4*>(h)[i] = v;
}

// ---------------------------------------------------------------------------
// Host-side orchestration
// ---------------------------------------------------------------------------
extern "C" void kernel_launch(void* const* d_in, const int* in_sizes, int n_in,
                              void* d_out, int out_size, void* d_ws, size_t ws_size,
                              hipStream_t stream) {
    (void)n_in; (void)out_size; (void)ws_size;
    const float* x        = (const float*)d_in[0];
    const int*   eidx     = (const int*)d_in[1];
    const float* lin1_w   = (const float*)d_in[2];
    const float* lin1_b   = (const float*)d_in[3];
    const float* conv_w1  = (const float*)d_in[4];
    const float* conv_w2  = (const float*)d_in[5];
    const float* gamma    = (const float*)d_in[6];
    const float* nbeta    = (const float*)d_in[7];
    const float* lin2_w   = (const float*)d_in[8];
    const float* lin2_b   = (const float*)d_in[9];

    const int DIN  = 128, DOUT = 128;
    const int N    = in_sizes[0] / DIN;           // 50000
    const int E    = in_sizes[1] / 2;             // 800000
    const int L    = in_sizes[4] / (DHID * DHID); // 8
    const int Mt   = N / 16;                      // 3125 (exact)
    const int* row = eidx;
    const int* col = eidx + E;

    char* p = (char*)d_ws;
    float*  x0   = (float*)p;  p += (size_t)N * DHID * 4;
    float*  h    = (float*)p;  p += (size_t)N * DHID * 4;
    float*  agg  = (float*)p;  p += (size_t)N * DHID * 4;
    float*  dis  = (float*)p;  p += (size_t)N * 4;
    float*  enorm= (float*)p;  p += (size_t)E * 4;
    __bf16* w1t  = (__bf16*)p; p += (size_t)L * DHID * DHID * 2;
    __bf16* w2t  = (__bf16*)p; p += (size_t)L * DHID * DHID * 2;
    __bf16* l1t  = (__bf16*)p; p += (size_t)DIN * DHID * 2;
    __bf16* l2t  = (__bf16*)p; p += (size_t)DHID * DOUT * 2;
    float*  stats= (float*)p;  p += 64;

    // --- graph normalization -------------------------------------------------
    hipMemsetAsync(dis, 0, (size_t)N * 4, stream);
    k_deg  <<<(E + 255) / 256, 256, 0, stream>>>(col, dis, E);
    k_dis  <<<(N + 255) / 256, 256, 0, stream>>>(dis, N);
    k_enorm<<<(E + 255) / 256, 256, 0, stream>>>(row, col, dis, enorm, E);

    // --- weight conversion (fp32 -> bf16 transposed) -------------------------
    {
        int tot = L * DHID * DHID;
        k_wt_bf16<<<(tot + 255) / 256, 256, 0, stream>>>(conv_w1, w1t, DHID, DHID, tot);
        k_wt_bf16<<<(tot + 255) / 256, 256, 0, stream>>>(conv_w2, w2t, DHID, DHID, tot);
        int t1 = DIN * DHID;
        k_wt_bf16<<<(t1 + 255) / 256, 256, 0, stream>>>(lin1_w, l1t, DIN, DHID, t1);
        int t2 = DHID * DOUT;
        k_wt_bf16<<<(t2 + 255) / 256, 256, 0, stream>>>(lin2_w, l2t, DHID, DOUT, t2);
    }

    // --- x0 = relu(x @ lin1_w + b) ------------------------------------------
    k_gemm_bias_relu<<<Mt, 32 * (DHID / 64), 0, stream>>>(x, l1t, lin1_b, x0, DIN, DHID);

    // --- GCNII layers --------------------------------------------------------
    for (int i = 0; i < L; ++i) {
        float bl = logf(1.0f / (float)(i + 1) + 1.0f);
        float c0 = (1.0f - bl) * 0.5f;
        float c1 = bl * 0.5f;
        const float* hsrc = (i == 0) ? x0 : h;

        hipMemsetAsync(agg,   0, (size_t)N * DHID * 4, stream);
        hipMemsetAsync(stats, 0, 16, stream);

        k_agg<<<(E + N + 7) / 8, 256, 0, stream>>>(row, col, enorm, dis, hsrc, agg, E, N);

        k_dualgemm_combine<<<Mt, 128, 0, stream>>>(
            agg, x0, w1t + (size_t)i * DHID * DHID, w2t + (size_t)i * DHID * DHID,
            h, stats, c0, c1);

        k_ln_finalize<<<1, 1, 0, stream>>>(stats, (float)N * (float)DHID);

        int total4 = N * DHID / 4;
        k_ln_apply<<<(total4 + 255) / 256, 256, 0, stream>>>(
            h, stats, gamma + (size_t)i * DHID, nbeta + (size_t)i * DHID, total4);
    }

    // --- out = relu(h @ lin2_w + b) ------------------------------------------
    k_gemm_bias_relu<<<Mt, 32 * (DOUT / 64), 0, stream>>>(h, l2t, lin2_b, (float*)d_out,
                                                          DHID, DOUT);
}
